// GLABlock_39711267619133
// MI455X (gfx1250) — compile-verified, matrix operand order checked by reading, not run on hardware
//
#include <hip/hip_runtime.h>

// ---------------------------------------------------------------------------
// GLA block forward for gfx1250 (MI455X).  bf16 WMMA + async LDS staging.
// All LDS fragment accesses are contiguous per-lane 2x16B runs (ds_load_b128).
// B=8 T=8192 D=512 H=4 DK=256 DV=512 HK=64 HV=128 C=64
// ---------------------------------------------------------------------------

typedef __attribute__((ext_vector_type(16))) __bf16 v16bf;
typedef __attribute__((ext_vector_type(8)))  float  v8f;

constexpr int Bc  = 8;
constexpr int Tt  = 8192;
constexpr int Dd  = 512;
constexpr int Hh  = 4;
constexpr int DKc = 256;
constexpr int DVc = 512;
constexpr int HKc = 64;
constexpr int HVc = 128;
constexpr int CC  = 64;
constexpr int NCH = Tt / CC;          // 128 chunks
constexpr float SCALE = 0.125f;       // HK^-0.5
constexpr float INV_GN = 1.0f / 16.0f;
constexpr float EPSc = 1e-5f;

static constexpr size_t Mrows = (size_t)Bc * Tt;   // 65536

// ---------------------------------------------------------------------------
// CDNA5 helpers
// ---------------------------------------------------------------------------
__device__ __forceinline__ v8f wmma_bf16(v16bf a, v16bf b, v8f c) {
  return __builtin_amdgcn_wmma_f32_16x16x32_bf16(false, a, false, b,
                                                 (short)0, c, false, false);
}

// low 32 bits of the generic address of a __shared__ object == LDS offset
__device__ __forceinline__ uint32_t lds_addr(const void* p) {
  return (uint32_t)(uintptr_t)p;
}

// async global->LDS copy of 16 bytes per lane (VGLOBAL GV mode, ASYNCcnt)
__device__ __forceinline__ void async_load_b128(uint32_t ldsOff, const void* g) {
  asm volatile("global_load_async_to_lds_b128 %0, %1, off"
               :: "v"(ldsOff), "v"((unsigned long long)(uintptr_t)g)
               : "memory");
}
__device__ __forceinline__ void wait_async0() {
  asm volatile("s_wait_asynccnt 0x0" ::: "memory");
}
__device__ __forceinline__ void wait_async4() {
  asm volatile("s_wait_asynccnt 0x4" ::: "memory");
}

// A/B 16x32 bf16 fragment: lanes 0-15 hold K {0..7,16..23} of row MN,
// lanes 16-31 hold K {8..15,24..31}.  Row-major K-fast source only ->
// two contiguous 16B runs per lane -> paired ds_load_b128.
__device__ __forceinline__ v16bf load_frag16(const __bf16* __restrict__ base,
                                             int ld) {
  int lane = threadIdx.x & 31;
  int mn = lane & 15, hi = lane >> 4;
  v16bf f;
#pragma unroll
  for (int j = 0; j < 16; ++j) {
    int k = (j < 8 ? j : j + 8) + hi * 8;
    f[j] = base[mn * ld + k];
  }
  return f;
}

// same, with per-lane-row scale folded in during conversion (k_dec trick)
__device__ __forceinline__ v16bf load_frag16_scaled(const __bf16* __restrict__ base,
                                                    int ld, float s) {
  int lane = threadIdx.x & 31;
  int mn = lane & 15, hi = lane >> 4;
  v16bf f;
#pragma unroll
  for (int j = 0; j < 16; ++j) {
    int k = (j < 8 ? j : j + 8) + hi * 8;
    f[j] = (__bf16)((float)base[mn * ld + k] * s);
  }
  return f;
}

// C/D 16x16 f32: VGPR r -> (M=r,N=lane) lanes0-15; (M=r+8,N=lane-16) lanes16-31
template <typename OutT>
__device__ __forceinline__ void store_cd_guard(OutT* __restrict__ C, size_t ldc,
                                               int m0, int n0, int M, int N, v8f acc) {
  int lane = threadIdx.x & 31, n = lane & 15, hi = lane >> 4;
#pragma unroll
  for (int r = 0; r < 8; ++r) {
    int m = m0 + r + hi * 8, nn = n0 + n;
    if (m < M && nn < N) C[(size_t)m * ldc + nn] = (OutT)acc[r];
  }
}

// transposed store: CT[n][m], per-lane 8 consecutive m -> contiguous bf16 run
__device__ __forceinline__ void store_cd_tr(__bf16* __restrict__ CT, size_t ldm,
                                            int m0, int n0, v8f acc) {
  int lane = threadIdx.x & 31, n = lane & 15, hi = lane >> 4;
  __bf16* p = CT + (size_t)(n0 + n) * ldm + m0 + hi * 8;
#pragma unroll
  for (int r = 0; r < 8; ++r) p[r] = (__bf16)acc[r];
}

// ---------------------------------------------------------------------------
// GEMM:  C[M,N] = A[M,K](bf16,rm) * W[N,K](bf16,rm)^T.  128x128 block tile,
// 8 waves x (2x4) 16x16 tiles, K-step 32, double-buffered async staging.
// ---------------------------------------------------------------------------
__device__ __forceinline__ void gemm_stage(const __bf16* __restrict__ Ag,
                                           const __bf16* __restrict__ Wg,
                                           int bm, int bn, int K, int k0,
                                           uint32_t aOff, uint32_t wOff, int tid) {
#pragma unroll
  for (int t = 0; t < 2; ++t) {
    int i = tid + t * 256;                 // 0..511 : 128 rows x 4 x 16B
    int r = i >> 2, s = i & 3;
    async_load_b128(aOff + (uint32_t)(r * 32 + s * 8) * 2,
                    Ag + (size_t)(bm + r) * K + k0 + s * 8);
    async_load_b128(wOff + (uint32_t)(r * 32 + s * 8) * 2,
                    Wg + (size_t)(bn + r) * K + k0 + s * 8);
  }
}

template <typename OutT, bool TR>
__global__ void __launch_bounds__(256)
gemm_async_kernel(const __bf16* __restrict__ A, const __bf16* __restrict__ W,
                  OutT* __restrict__ C, int M, int N, int K) {
  __shared__ __bf16 AsBuf[2][128 * 32];
  __shared__ __bf16 WsBuf[2][128 * 32];
  const int bm = blockIdx.x * 128, bn = blockIdx.y * 128;
  const int tid = threadIdx.x, wave = tid >> 5;
  const int wtm = wave & 3;              // m-tile pair: rows 32*wtm..+32
  const int wtn = (wave >> 2) * 4;       // n tiles wtn..wtn+3
  const uint32_t aOff[2] = {lds_addr(AsBuf[0]), lds_addr(AsBuf[1])};
  const uint32_t wOff[2] = {lds_addr(WsBuf[0]), lds_addr(WsBuf[1])};

  v8f acc0[4] = {}, acc1[4] = {};
  const int NK = K >> 5;

  gemm_stage(A, W, bm, bn, K, 0, aOff[0], wOff[0], tid);
  for (int it = 0; it < NK; ++it) {
    const int cur = it & 1;
    if (it + 1 < NK) {
      gemm_stage(A, W, bm, bn, K, (it + 1) * 32, aOff[cur ^ 1], wOff[cur ^ 1], tid);
      wait_async4();                     // stage `it` landed (in-order)
    } else {
      wait_async0();
    }
    __syncthreads();
    const __bf16* As = AsBuf[cur];
    const __bf16* Ws = WsBuf[cur];
    v16bf a0 = load_frag16(As + (wtm * 2 + 0) * 16 * 32, 32);
    v16bf a1 = load_frag16(As + (wtm * 2 + 1) * 16 * 32, 32);
#pragma unroll
    for (int j = 0; j < 4; ++j) {
      v16bf bb = load_frag16(Ws + (wtn + j) * 16 * 32, 32);
      acc0[j] = wmma_bf16(a0, bb, acc0[j]);
      acc1[j] = wmma_bf16(a1, bb, acc1[j]);
    }
    __syncthreads();
  }
#pragma unroll
  for (int j = 0; j < 4; ++j) {
    if constexpr (TR) {
      store_cd_tr((__bf16*)C, (size_t)M, bm + (wtm * 2 + 0) * 16, bn + (wtn + j) * 16, acc0[j]);
      store_cd_tr((__bf16*)C, (size_t)M, bm + (wtm * 2 + 1) * 16, bn + (wtn + j) * 16, acc1[j]);
    } else {
      store_cd_guard(C, (size_t)N, bm + (wtm * 2 + 0) * 16, bn + (wtn + j) * 16, M, N, acc0[j]);
      store_cd_guard(C, (size_t)N, bm + (wtm * 2 + 1) * 16, bn + (wtn + j) * 16, M, N, acc1[j]);
    }
  }
}

// ---------------------------------------------------------------------------
// Elementwise / small kernels
// ---------------------------------------------------------------------------
__global__ void f32_to_bf16_fill_kernel(const float* __restrict__ src,
                                        __bf16* __restrict__ dst,
                                        size_t n_src, size_t n_total) {
  size_t i = (size_t)blockIdx.x * blockDim.x + threadIdx.x;
  if (i < n_total) dst[i] = (i < n_src) ? (__bf16)src[i] : (__bf16)0.0f;
}

__global__ void f32_to_bf16_pad_kernel(const float* __restrict__ src,
                                       __bf16* __restrict__ dst,
                                       size_t rows, int sc, int dc) {
  size_t i = (size_t)blockIdx.x * blockDim.x + threadIdx.x;
  if (i >= rows * (size_t)dc) return;
  size_t r = i / dc;
  int c = (int)(i % dc);
  dst[i] = (c < sc) ? (__bf16)src[r * sc + c] : (__bf16)0.0f;
}

__global__ void gate_logsig_kernel(float* __restrict__ h2,
                                   const float* __restrict__ bias, size_t n) {
  size_t i = (size_t)blockIdx.x * blockDim.x + threadIdx.x;
  if (i >= n) return;
  float x = h2[i] + bias[i % DKc];
  float ls = fminf(x, 0.0f) - log1pf(__expf(-fabsf(x)));
  h2[i] = ls * INV_GN;
}

__global__ void __launch_bounds__(256)
chunk_cumsum_kernel(float* __restrict__ gk) {
  int blk = blockIdx.x;
  int b = blk / NCH, ch = blk % NCH;
  int col = threadIdx.x;
  size_t base = ((size_t)b * Tt + (size_t)ch * CC) * DKc + col;
  float acc = 0.0f;
  for (int r = 0; r < CC; ++r) {
    acc += gk[base + (size_t)r * DKc];
    gk[base + (size_t)r * DKc] = acc;
  }
}

// qe = q*exp(bc)*scale, ke = k*exp(-bc) (token-major) + keT (d-major, per b,h)
__global__ void prep_qk_kernel(const float* __restrict__ q,
                               const float* __restrict__ k,
                               const float* __restrict__ bc,
                               __bf16* __restrict__ qe, __bf16* __restrict__ ke,
                               __bf16* __restrict__ keT, size_t n) {
  size_t i = (size_t)blockIdx.x * blockDim.x + threadIdx.x;
  if (i >= n) return;
  float b = bc[i];
  float kev = k[i] * __expf(-b);
  qe[i] = (__bf16)(q[i] * __expf(b) * SCALE);
  ke[i] = (__bf16)kev;
  size_t m = i / DKc;            // global token row
  int col = (int)(i % DKc);
  size_t bb = m >> 13;           // /Tt
  size_t t  = m & (Tt - 1);
  int h = col >> 6, d = col & 63;
  keT[((bb * Hh + h) * HKc + d) * Tt + t] = (__bf16)kev;
}

// ---------------------------------------------------------------------------
// Chunked GLA scan.  One workgroup per (b,h); 8 waves.  State S (64x128 f32)
// register-resident (4 C/D tiles per wave) with transposed bf16 LDS mirror.
// All chunk operands arrive via global_load_async_to_lds_b128; every LDS
// fragment read/write is contiguous per lane.
// ---------------------------------------------------------------------------
__global__ void __launch_bounds__(256)
gla_scan_kernel(const __bf16* __restrict__ qeg, const __bf16* __restrict__ keg,
                const __bf16* __restrict__ keTg, const __bf16* __restrict__ vTg,
                const float* __restrict__ bcg, float* __restrict__ og) {
  extern __shared__ __align__(16) char smem[];
  __bf16* SbT = (__bf16*)(smem);              // [HVc][HKc] 16 KB state mirror
  __bf16* vbT = (__bf16*)(smem + 16384);      // [HVc][CC]  16 KB
  __bf16* qe  = (__bf16*)(smem + 32768);      // [CC][HKc]   8 KB
  __bf16* ke  = (__bf16*)(smem + 40960);      // [CC][HKc]   8 KB
  __bf16* keT = (__bf16*)(smem + 49152);      // [HKc][CC]   8 KB
  __bf16* Ab  = (__bf16*)(smem + 57344);      // [CC][CC]    8 KB (A row-major)
  float* blast = (float*)(smem + 65536);      // 256 B

  const int b = blockIdx.x / Hh;
  const int h = blockIdx.x % Hh;
  const int tid = threadIdx.x;
  const int wave = tid >> 5;
  const int lane = tid & 31, ln = lane & 15, lhi = lane >> 4;

  const size_t baseQK  = (size_t)b * Tt * DKc + (size_t)h * HKc;
  const size_t baseV   = (size_t)b * Tt * DVc + (size_t)h * HVc;
  const size_t baseKeT = ((size_t)(b * Hh + h) * HKc) * Tt;        // + d*Tt + t
  const size_t baseVT  = (size_t)h * HVc * Mrows + (size_t)b * Tt; // + v*M + t

  const uint32_t qeOff = lds_addr(qe);
  const uint32_t keOff = lds_addr(ke);
  const uint32_t keTOff = lds_addr(keT);
  const uint32_t vbTOff = lds_addr(vbT);

  const int smt = wave & 3, sntb = (wave >> 2) * 4;
  v8f Sreg[4] = {};

  for (int i = tid; i < HVc * HKc; i += 256) SbT[i] = (__bf16)0.0f;
  __syncthreads();

  for (int ch = 0; ch < NCH; ++ch) {
    const int t0 = ch * CC;

    // ---- async stage: qe/ke (64x64), keT (64x64), vbT (128x64) ----
#pragma unroll
    for (int t = 0; t < 2; ++t) {
      int i = tid + t * 256;                 // 0..511, 64 rows x 8 x 16B
      int r = i >> 3, s = i & 7;
      uint32_t l = (uint32_t)(r * 64 + s * 8) * 2;
      size_t gqk = baseQK + (size_t)(t0 + r) * DKc + s * 8;
      async_load_b128(qeOff + l, qeg + gqk);
      async_load_b128(keOff + l, keg + gqk);
      async_load_b128(keTOff + l, keTg + baseKeT + (size_t)r * Tt + t0 + s * 8);
    }
#pragma unroll
    for (int t = 0; t < 4; ++t) {
      int i = tid + t * 256;                 // 0..1023, 128 rows x 8 x 16B
      int r = i >> 3, s = i & 7;
      async_load_b128(vbTOff + (uint32_t)(r * 64 + s * 8) * 2,
                      vTg + baseVT + (size_t)r * Mrows + t0 + s * 8);
    }
    if (tid < HKc)
      blast[tid] = bcg[baseQK + (size_t)(t0 + CC - 1) * DKc + tid];
    wait_async0();
    __syncthreads();

    // ---- phase 1: AT = ke * qe^T (both token-major), store transposed ----
    // A[i][j] = AT[j][i]; keep i >= j (tril in A).
    {
      int t = wave * 2;
      for (int s = 0; s < 2; ++s, ++t) {
        int mt = t >> 2, nt = t & 3;         // mt: j tiles, nt: i tiles
        v8f acc = {};
        if (nt >= mt) {
#pragma unroll
          for (int ks = 0; ks < 2; ++ks) {
            v16bf a  = load_frag16(ke + mt * 16 * HKc + ks * 32, HKc);
            v16bf bb = load_frag16(qe + nt * 16 * HKc + ks * 32, HKc);
            acc = wmma_bf16(a, bb, acc);
          }
        }
        // element (m=j, n=i) -> Ab[i*CC + j], contiguous 8-run per lane
        __bf16* p = Ab + (nt * 16 + ln) * CC + mt * 16 + lhi * 8;
        int iG = nt * 16 + ln, jG = mt * 16 + lhi * 8;
#pragma unroll
        for (int r = 0; r < 8; ++r)
          p[r] = (__bf16)((iG >= jG + r) ? acc[r] : 0.0f);
      }
    }
    __syncthreads();

    // ---- phase 2: o = Ab*v + qe*S_old   (4 tiles / wave, 64x128) ----
    {
#pragma unroll
      for (int j = 0; j < 4; ++j) {
        int nt = sntb + j;
        v8f acc = {};
#pragma unroll
        for (int ks = 0; ks < 2; ++ks) {
          v16bf a  = load_frag16(Ab + smt * 16 * CC + ks * 32, CC);
          v16bf bb = load_frag16(vbT + nt * 16 * CC + ks * 32, CC);
          acc = wmma_bf16(a, bb, acc);
        }
#pragma unroll
        for (int ks = 0; ks < 2; ++ks) {
          v16bf a  = load_frag16(qe + smt * 16 * HKc + ks * 32, HKc);
          v16bf bb = load_frag16(SbT + nt * 16 * HKc + ks * 32, HKc);
          acc = wmma_bf16(a, bb, acc);
        }
        size_t ob = baseV + (size_t)(t0 + smt * 16) * DVc + nt * 16;
#pragma unroll
        for (int r = 0; r < 8; ++r)
          og[ob + (size_t)(r + lhi * 8) * DVc + ln] = acc[r];
      }
    }
    __syncthreads();

    // ---- phase 3: S = diag(exp(b_last)) * S + k_dec^T * v ----
    // k_dec^T[d][c] = keT[d][c]*exp(b_last[d]) -> per-lane-row constant.
    {
      float efrow = __expf(blast[smt * 16 + ln]);
#pragma unroll
      for (int j = 0; j < 4; ++j) {
        int nt = sntb + j;
        v8f acc;
#pragma unroll
        for (int r = 0; r < 8; ++r) {
          int m = smt * 16 + r + lhi * 8;
          acc[r] = Sreg[j][r] * __expf(blast[m]);
        }
#pragma unroll
        for (int ks = 0; ks < 2; ++ks) {
          v16bf a  = load_frag16_scaled(keT + smt * 16 * CC + ks * 32, CC, efrow);
          v16bf bb = load_frag16(vbT + nt * 16 * CC + ks * 32, CC);
          acc = wmma_bf16(a, bb, acc);
        }
        Sreg[j] = acc;
        // SbT[v][d]: per lane v=nt*16+ln fixed, 8 consecutive d -> b128 run
        __bf16* p = SbT + (nt * 16 + ln) * HKc + smt * 16 + lhi * 8;
#pragma unroll
        for (int r = 0; r < 8; ++r) p[r] = (__bf16)acc[r];
      }
    }
    __syncthreads();
  }
}

// ---------------------------------------------------------------------------
// RMS-norm over head dim + SiLU(g) gate -> bf16 for the final Wo GEMM.
// ---------------------------------------------------------------------------
__global__ void __launch_bounds__(128)
gate_norm_kernel(const float* __restrict__ o, const float* __restrict__ g,
                 const float* __restrict__ gnw, __bf16* __restrict__ ohat) {
  __shared__ float red[128];
  size_t blk = blockIdx.x;
  int hh = (int)(blk % Hh);
  size_t tok = blk / Hh;
  size_t base = tok * DVc + (size_t)hh * HVc;
  int c = threadIdx.x;
  float ov = o[base + c];
  red[c] = ov * ov;
  __syncthreads();
  for (int s = 64; s > 0; s >>= 1) {
    if (c < s) red[c] += red[c + s];
    __syncthreads();
  }
  float rms = rsqrtf(red[0] * (1.0f / HVc) + EPSc);
  float gv = g[base + c];
  float silu = gv / (1.0f + __expf(-gv));
  ohat[base + c] = (__bf16)(ov * rms * gnw[c] * silu);
}

// ---------------------------------------------------------------------------
// launcher
// ---------------------------------------------------------------------------
static inline size_t aln(size_t x) { return (x + 255) & ~(size_t)255; }

extern "C" void kernel_launch(void* const* d_in, const int* in_sizes, int n_in,
                              void* d_out, int out_size, void* d_ws, size_t ws_size,
                              hipStream_t stream) {
  (void)in_sizes; (void)n_in; (void)out_size; (void)ws_size;
  const float* x    = (const float*)d_in[0];
  const float* Wq   = (const float*)d_in[1];
  const float* Wk   = (const float*)d_in[2];
  const float* Wv   = (const float*)d_in[3];
  const float* Wg   = (const float*)d_in[4];
  const float* Wgk1 = (const float*)d_in[5];
  const float* Wgk2 = (const float*)d_in[6];
  const float* bgk2 = (const float*)d_in[7];
  const float* gnw  = (const float*)d_in[8];
  const float* Wo   = (const float*)d_in[9];
  float* out = (float*)d_out;

  char* p = (char*)d_ws;
  auto take = [&](size_t bytes) { char* q = p; p += aln(bytes); return q; };

  __bf16* xb     = (__bf16*)take(Mrows * Dd * 2);
  __bf16* wq_b   = (__bf16*)take((size_t)DKc * Dd * 2);
  __bf16* wk_b   = (__bf16*)take((size_t)DKc * Dd * 2);
  __bf16* wv_b   = (__bf16*)take((size_t)DVc * Dd * 2);
  __bf16* wg_b   = (__bf16*)take((size_t)DVc * Dd * 2);
  __bf16* wo_b   = (__bf16*)take((size_t)Dd * Dd * 2);
  __bf16* wgk1_b = (__bf16*)take((size_t)128 * Dd * 2);   // 16 rows padded to 128
  __bf16* wgk2_b = (__bf16*)take((size_t)DKc * 32 * 2);   // K 16 padded to 32
  float*  q_f  = (float*)take(Mrows * DKc * 4);
  float*  k_f  = (float*)take(Mrows * DKc * 4);
  __bf16* vT_b = (__bf16*)take(Mrows * DVc * 2);          // [DV][Mrows]
  float*  g_f  = (float*)take(Mrows * DVc * 4);
  float*  hlow = (float*)take(Mrows * 16 * 4);
  __bf16* hlow_b = (__bf16*)take(Mrows * 32 * 2);
  float*  gk_f = (float*)take(Mrows * DKc * 4);           // becomes cumsum bc
  __bf16* qe_b = (__bf16*)take(Mrows * DKc * 2);
  __bf16* ke_b = (__bf16*)take(Mrows * DKc * 2);
  __bf16* keT_b = (__bf16*)take(Mrows * DKc * 2);         // [B*H*HK][T]
  float*  o_f  = (float*)take(Mrows * DVc * 4);
  __bf16* ohat = (__bf16*)take(Mrows * DVc * 2);

  auto cvt = [&](const float* s, __bf16* d, size_t n_src, size_t n_tot) {
    f32_to_bf16_fill_kernel<<<dim3((unsigned)((n_tot + 255) / 256)), 256, 0, stream>>>(
        s, d, n_src, n_tot);
  };
  cvt(x, xb, Mrows * Dd, Mrows * Dd);
  cvt(Wq, wq_b, (size_t)DKc * Dd, (size_t)DKc * Dd);
  cvt(Wk, wk_b, (size_t)DKc * Dd, (size_t)DKc * Dd);
  cvt(Wv, wv_b, (size_t)DVc * Dd, (size_t)DVc * Dd);
  cvt(Wg, wg_b, (size_t)DVc * Dd, (size_t)DVc * Dd);
  cvt(Wo, wo_b, (size_t)Dd * Dd, (size_t)Dd * Dd);
  cvt(Wgk1, wgk1_b, (size_t)16 * Dd, (size_t)128 * Dd);   // zero-pad rows 16..127
  {
    size_t n = (size_t)DKc * 32;
    f32_to_bf16_pad_kernel<<<dim3((unsigned)((n + 255) / 256)), 256, 0, stream>>>(
        Wgk2, wgk2_b, DKc, 16, 32);
  }

  auto gemm_f32 = [&](const __bf16* A, const __bf16* W, float* C, int M, int N, int K) {
    dim3 grid((unsigned)((M + 127) / 128), (unsigned)((N + 127) / 128));
    gemm_async_kernel<float, false><<<grid, 256, 0, stream>>>(A, W, C, M, N, K);
  };
  auto gemm_b16tr = [&](const __bf16* A, const __bf16* W, __bf16* C, int M, int N, int K) {
    dim3 grid((unsigned)((M + 127) / 128), (unsigned)((N + 127) / 128));
    gemm_async_kernel<__bf16, true><<<grid, 256, 0, stream>>>(A, W, C, M, N, K);
  };

  gemm_f32(xb, wq_b, q_f, (int)Mrows, DKc, Dd);
  gemm_f32(xb, wk_b, k_f, (int)Mrows, DKc, Dd);
  gemm_b16tr(xb, wv_b, vT_b, (int)Mrows, DVc, Dd);    // V stored transposed
  gemm_f32(xb, wg_b, g_f, (int)Mrows, DVc, Dd);
  gemm_f32(xb, wgk1_b, hlow, (int)Mrows, 16, Dd);     // W rows padded to 128

  {
    size_t n = Mrows * 32;
    f32_to_bf16_pad_kernel<<<dim3((unsigned)((n + 255) / 256)), 256, 0, stream>>>(
        hlow, hlow_b, Mrows, 16, 32);
  }
  gemm_f32(hlow_b, wgk2_b, gk_f, (int)Mrows, DKc, 32);

  {
    size_t n = Mrows * DKc;
    gate_logsig_kernel<<<dim3((unsigned)((n + 255) / 256)), 256, 0, stream>>>(
        gk_f, bgk2, n);
  }
  chunk_cumsum_kernel<<<dim3((unsigned)(Bc * NCH)), 256, 0, stream>>>(gk_f);

  {
    size_t n = Mrows * DKc;
    prep_qk_kernel<<<dim3((unsigned)((n + 255) / 256)), 256, 0, stream>>>(
        q_f, k_f, gk_f, qe_b, ke_b, keT_b, n);
  }

  gla_scan_kernel<<<dim3((unsigned)(Bc * Hh)), 256, 65792, stream>>>(
      qe_b, ke_b, keT_b, vT_b, gk_f, o_f);

  gate_norm_kernel<<<dim3((unsigned)(Mrows * Hh)), 128, 0, stream>>>(
      o_f, g_f, gnw, ohat);

  gemm_f32(ohat, wo_b, out, (int)Mrows, Dd, Dd);
}
